// MixerLayer_46205258170887
// MI455X (gfx1250) — matrix operand order repaired and checked by reading, not compiled, not run on hardware
//
#include <hip/hip_runtime.h>
#include <hip/hip_bf16.h>

typedef __bf16 bf16_t;
typedef __bf16 v8bf  __attribute__((ext_vector_type(8)));
typedef __bf16 v16bf __attribute__((ext_vector_type(16)));
typedef float  v8f   __attribute__((ext_vector_type(8)));

#define EMB   256
#define SEQ   1024
#define BATCH 16
#define NH    8
#define DH    32
#define IMH   16
#define IMW   64
#define NEG_BIG (-1.0e30f)

#if defined(__has_builtin)
#if __has_builtin(__builtin_amdgcn_sched_barrier)
#define SCHED_FENCE() __builtin_amdgcn_sched_barrier(0)
#endif
#endif
#ifndef SCHED_FENCE
#define SCHED_FENCE()
#endif

static __device__ __forceinline__ v16bf cat16(v8bf a, v8bf b) {
  v16bf r;
#pragma unroll
  for (int i = 0; i < 8; ++i) { r[i] = a[i]; r[8 + i] = b[i]; }
  return r;
}

static __device__ __forceinline__ v8f wmma_bf16(v16bf a, v16bf b, v8f c) {
  return __builtin_amdgcn_wmma_f32_16x16x32_bf16(false, a, false, b,
                                                 (short)0, c, false, false);
}

// ---------------- prep kernels ----------------

// vectorized f32 -> bf16: 8 elements per thread
__global__ void k_cvt_bf16(const float* __restrict__ x, bf16_t* __restrict__ y, int n8) {
  int i = blockIdx.x * blockDim.x + threadIdx.x;
  if (i < n8) {
    const float* p = x + (long)i * 8;
    v8bf o;
#pragma unroll
    for (int j = 0; j < 8; ++j) o[j] = (bf16_t)p[j];
    *(v8bf*)(y + (long)i * 8) = o;
  }
}

// in_proj_w [3E][E] (out,in) -> Wqkv_t [E][3E] bf16 so B-tiles are contiguous
__global__ void k_transpose_wqkv(const float* __restrict__ w, bf16_t* __restrict__ wt) {
  int idx = blockIdx.x * 256 + threadIdx.x;          // 768*256
  int o = idx / 256, e = idx % 256;
  wt[(long)e * 768 + o] = (bf16_t)w[(long)o * 256 + e];
}

// Wc[o][e] = sum_m proj_w[o][m] * out_proj_w[m][e], stored transposed [e][o] bf16
__global__ void k_combine_w(const float* __restrict__ pw, const float* __restrict__ ow,
                            bf16_t* __restrict__ wct) {
  int idx = blockIdx.x * 256 + threadIdx.x;          // 256*256
  int o = idx >> 8, e = idx & 255;
  float acc = 0.f;
  for (int m = 0; m < 256; ++m) acc += pw[o * 256 + m] * ow[m * 256 + e];
  wct[(long)e * 256 + o] = (bf16_t)acc;
}

// bc[o] = sum_e proj_w[o][e] * out_proj_b[e]
__global__ void k_combine_b(const float* __restrict__ pw, const float* __restrict__ ob,
                            float* __restrict__ bc) {
  int o = threadIdx.x;
  float acc = 0.f;
  for (int e = 0; e < 256; ++e) acc += pw[o * 256 + e] * ob[e];
  bc[o] = acc;
}

// ---------------- QKV projection: [16384,256] @ [256,768] + b ----------------
// 16x64 strip per wave: one A tile feeds 4 WMMAs per K-step. Next-step loads
// are issued BEFORE a sched_barrier so they cannot be sunk below the current
// step's WMMAs -> WMMA latency covers the L2 round trip.

__global__ __launch_bounds__(256, 1)
void k_qkv_gemm(const bf16_t* __restrict__ Xb,
                const bf16_t* __restrict__ Wt,   // [256][768]
                const float* __restrict__ bias,  // [768]
                bf16_t* __restrict__ QKV) {      // [16384][768]
  const int lane = threadIdx.x & 31;
  const int wv   = threadIdx.x >> 5;
  const int wave = blockIdx.x * 8 + wv;   // 1024 * 12 waves
  const int mt = wave / 12, ng = wave % 12;
  const int m0 = mt * 16, n0 = ng * 64;
  const int r = lane & 15, hf = lane >> 4;

  v8f acc[4] = {};
  const bf16_t* ap = Xb + (long)(m0 + r) * EMB;

  // prologue: load K-step 0
  v16bf A = cat16(*(const v8bf*)(ap + 8 * hf),
                  *(const v8bf*)(ap + 16 + 8 * hf));
  v16bf Bm[4];
  {
    const bf16_t* bp = Wt + n0 + r;
#pragma unroll
    for (int t = 0; t < 4; ++t)
#pragma unroll
      for (int s = 0; s < 16; ++s)
        Bm[t][s] = bp[(long)(s < 8 ? 8 * hf + s : 8 + 8 * hf + s) * 768 + 16 * t];
  }

#pragma unroll
  for (int k0 = 0; k0 < EMB; k0 += 32) {
    v16bf An, Bn[4];
    const bool more = (k0 + 32 < EMB);
    if (more) {   // prefetch next step
      An = cat16(*(const v8bf*)(ap + k0 + 32 + 8 * hf),
                 *(const v8bf*)(ap + k0 + 48 + 8 * hf));
      const bf16_t* bp = Wt + (long)(k0 + 32) * 768 + n0 + r;
#pragma unroll
      for (int t = 0; t < 4; ++t)
#pragma unroll
        for (int s = 0; s < 16; ++s)
          Bn[t][s] = bp[(long)(s < 8 ? 8 * hf + s : 8 + 8 * hf + s) * 768 + 16 * t];
    }
    SCHED_FENCE();   // keep next-step loads above this step's WMMAs
#pragma unroll
    for (int t = 0; t < 4; ++t) acc[t] = wmma_bf16(A, Bm[t], acc[t]);
    SCHED_FENCE();
    if (more) {
      A = An;
#pragma unroll
      for (int t = 0; t < 4; ++t) Bm[t] = Bn[t];
    }
  }

#pragma unroll
  for (int t = 0; t < 4; ++t) {
    const float bv = bias[n0 + 16 * t + r];
#pragma unroll
    for (int v = 0; v < 8; ++v)
      QKV[(long)(m0 + v + 8 * hf) * 768 + n0 + 16 * t + r] = (bf16_t)(acc[t][v] + bv);
  }
}

// ---------------- banded flash attention ----------------
// one wave per 16-query tile (one image-row segment); S^T = K@Q^T so the f32
// D-layout of S^T doubles as the bf16 A-layout of P for P@V (no LDS transpose).

__global__ void k_attn(const bf16_t* __restrict__ QKV, bf16_t* __restrict__ CTX) {
  const int lane = threadIdx.x & 31;
  const int wv   = threadIdx.x >> 5;
  const int blk  = blockIdx.x;            // 16 * 8 * 8
  const int b  = blk >> 6;
  const int h  = (blk >> 3) & 7;
  const int qg = blk & 7;
  const int qbase = (qg * 8 + wv) * 16;
  const int row  = qbase >> 6;            // image row of this query tile
  const int colb = qbase & 63;
  const int r = lane & 15, hf = lane >> 4;

  const long base = (long)b * SEQ * 768;
  const bf16_t* Q = QKV + base + (long)h * DH;
  const bf16_t* K = QKV + base + 256 + (long)h * DH;
  const bf16_t* V = QKV + base + 512 + (long)h * DH;

  // Q^T in B-layout: lane <-> query qbase+r, element <-> d (table order)
  v16bf qB = cat16(*(const v8bf*)(Q + (long)(qbase + r) * 768 + 8 * hf),
                   *(const v8bf*)(Q + (long)(qbase + r) * 768 + 16 + 8 * hf));

  v8f o1 = {}, o2 = {};
  float m_i = NEG_BIG, l_i = 0.f;
  const float scale = 0.17677669529663687f;  // 1/sqrt(32)
  const int c_q = colb + r;                  // this lane's query column

  for (int dr = -3; dr <= 3; ++dr) {
    const int r2 = row + dr;
    if (r2 < 0 || r2 >= IMH) continue;
#pragma unroll
    for (int cp = 0; cp < 2; ++cp) {
      const int key0 = r2 * 64 + cp * 32;   // 32 keys: two 16-wide col tiles
      // K tiles in A-layout (M = key, K = d)
      const long kr0 = (long)(key0 + r) * 768;
      const long kr1 = (long)(key0 + 16 + r) * 768;
      v16bf A0 = cat16(*(const v8bf*)(K + kr0 + 8 * hf),
                       *(const v8bf*)(K + kr0 + 16 + 8 * hf));
      v16bf A1 = cat16(*(const v8bf*)(K + kr1 + 8 * hf),
                       *(const v8bf*)(K + kr1 + 16 + 8 * hf));
      // V pair in B-layout: lane <-> d, element <-> key (table order) (issue early;
      // the fence keeps these loads above the S WMMAs + softmax VALU)
      v16bf B1, B2;
#pragma unroll
      for (int s = 0; s < 16; ++s) {
        const long vr = (long)(key0 + (s < 8 ? 8 * hf + s : 8 + 8 * hf + s)) * 768;
        B1[s] = V[vr + r];
        B2[s] = V[vr + 16 + r];
      }
      SCHED_FENCE();

      v8f s0 = {}, s1 = {};
      s0 = wmma_bf16(A0, qB, s0);           // S^T: M=key_local, N=query
      s1 = wmma_bf16(A1, qB, s1);

      float mloc = NEG_BIG;
#pragma unroll
      for (int v = 0; v < 8; ++v) {
        const int c2a = cp * 32 + v + 8 * hf;   // key column, tile 0
        const int c2b = c2a + 16;               // key column, tile 1
        const int da = c2a - c_q, db = c2b - c_q;
        float sa = (da >= -5 && da <= 5) ? s0[v] * scale : NEG_BIG;
        float sb = (db >= -5 && db <= 5) ? s1[v] * scale : NEG_BIG;
        s0[v] = sa; s1[v] = sb;
        mloc = fmaxf(mloc, fmaxf(sa, sb));
      }
      mloc = fmaxf(mloc, __shfl_xor(mloc, 16));   // combine key halves
      const float m_new = fmaxf(m_i, mloc);
      const float alpha = __expf(m_i - m_new);    // finite mask => no NaN

      float rsum = 0.f;
      v16bf P;  // exactly the A-layout of P (16q x 32k) — no shuffle needed
#pragma unroll
      for (int v = 0; v < 8; ++v) {
        float pa = __expf(s0[v] - m_new);
        float pb = __expf(s1[v] - m_new);
        rsum += pa + pb;
        P[v] = (bf16_t)pa;
        P[8 + v] = (bf16_t)pb;
      }
      rsum += __shfl_xor(rsum, 16);
      l_i = l_i * alpha + rsum;
      m_i = m_new;

#pragma unroll
      for (int v = 0; v < 8; ++v) {        // rescale O by alpha of row q=v+8*hf
        float av = __shfl(alpha, v + 8 * hf);
        o1[v] *= av; o2[v] *= av;
      }
      o1 = wmma_bf16(P, B1, o1);           // O: M=query, N=d0..15
      o2 = wmma_bf16(P, B2, o2);           //            N=d16..31
    }
  }

#pragma unroll
  for (int v = 0; v < 8; ++v) {
    float lq  = __shfl(l_i, v + 8 * hf);
    float inv = 1.0f / lq;                 // self-key guarantees lq >= ~1
    long orow = ((long)b * SEQ + qbase + v + 8 * hf) * EMB + (long)h * DH;
    CTX[orow + r]      = (bf16_t)(o1[v] * inv);
    CTX[orow + 16 + r] = (bf16_t)(o2[v] * inv);
  }
}

// ---------------- fused out_proj+proj: [16384,256] @ Wc^T + bc -> f32 ----------------
// 16x64 strip per wave, same fenced double-buffered structure as k_qkv_gemm

__global__ __launch_bounds__(256, 1)
void k_out_gemm(const bf16_t* __restrict__ Cb,
                const bf16_t* __restrict__ Wct,  // [256][256]
                const float* __restrict__ bc,
                float* __restrict__ out) {
  const int lane = threadIdx.x & 31;
  const int wv   = threadIdx.x >> 5;
  const int wave = blockIdx.x * 8 + wv;   // 1024 * 4 waves
  const int mt = wave >> 2, ng = wave & 3;
  const int m0 = mt * 16, n0 = ng * 64;
  const int r = lane & 15, hf = lane >> 4;

  v8f acc[4] = {};
  const bf16_t* ap = Cb + (long)(m0 + r) * EMB;

  v16bf A = cat16(*(const v8bf*)(ap + 8 * hf),
                  *(const v8bf*)(ap + 16 + 8 * hf));
  v16bf Bm[4];
  {
    const bf16_t* bp = Wct + n0 + r;
#pragma unroll
    for (int t = 0; t < 4; ++t)
#pragma unroll
      for (int s = 0; s < 16; ++s)
        Bm[t][s] = bp[(long)(s < 8 ? 8 * hf + s : 8 + 8 * hf + s) * 256 + 16 * t];
  }

#pragma unroll
  for (int k0 = 0; k0 < EMB; k0 += 32) {
    v16bf An, Bn[4];
    const bool more = (k0 + 32 < EMB);
    if (more) {
      An = cat16(*(const v8bf*)(ap + k0 + 32 + 8 * hf),
                 *(const v8bf*)(ap + k0 + 48 + 8 * hf));
      const bf16_t* bp = Wct + (long)(k0 + 32) * 256 + n0 + r;
#pragma unroll
      for (int t = 0; t < 4; ++t)
#pragma unroll
        for (int s = 0; s < 16; ++s)
          Bn[t][s] = bp[(long)(s < 8 ? 8 * hf + s : 8 + 8 * hf + s) * 256 + 16 * t];
    }
    SCHED_FENCE();
#pragma unroll
    for (int t = 0; t < 4; ++t) acc[t] = wmma_bf16(A, Bm[t], acc[t]);
    SCHED_FENCE();
    if (more) {
      A = An;
#pragma unroll
      for (int t = 0; t < 4; ++t) Bm[t] = Bn[t];
    }
  }

#pragma unroll
  for (int t = 0; t < 4; ++t) {
    const float bv = bc[n0 + 16 * t + r];
#pragma unroll
    for (int v = 0; v < 8; ++v)
      out[(long)(m0 + v + 8 * hf) * EMB + n0 + 16 * t + r] = acc[t][v] + bv;
  }
}

// ---------------- launch ----------------

extern "C" void kernel_launch(void* const* d_in, const int* in_sizes, int n_in,
                              void* d_out, int out_size, void* d_ws, size_t ws_size,
                              hipStream_t stream) {
  (void)in_sizes; (void)n_in; (void)out_size; (void)ws_size;
  const float* x     = (const float*)d_in[0];  // patch_embed [16,1024,256]
  const float* wqkv  = (const float*)d_in[1];  // in_proj_w [768,256]
  const float* bqkv  = (const float*)d_in[2];  // in_proj_b [768]
  const float* wout  = (const float*)d_in[3];  // out_proj_w [256,256]
  const float* bout  = (const float*)d_in[4];  // out_proj_b [256]
  const float* wproj = (const float*)d_in[5];  // proj_w [256,256]
  float* out = (float*)d_out;

  char* ws = (char*)d_ws;                      // ~40.5 MB total
  bf16_t* Xb   = (bf16_t*)(ws);                //  8,388,608 B
  bf16_t* Wqkt = (bf16_t*)(ws +  8388608);     //    393,216 B
  bf16_t* Wct  = (bf16_t*)(ws +  8781824);     //    131,072 B
  float*  bc   = (float*) (ws +  8912896);     //      1,024 B
  bf16_t* QKV  = (bf16_t*)(ws +  8913920);     // 25,165,824 B
  bf16_t* CTX  = (bf16_t*)(ws + 34079744);     //  8,388,608 B

  k_cvt_bf16      <<<2048, 256, 0, stream>>>(x, Xb, BATCH * SEQ * EMB / 8);
  k_transpose_wqkv<<< 768, 256, 0, stream>>>(wqkv, Wqkt);
  k_combine_w     <<< 256, 256, 0, stream>>>(wproj, wout, Wct);
  k_combine_b     <<<   1, 256, 0, stream>>>(wproj, bout, bc);
  k_qkv_gemm      <<<1536, 256, 0, stream>>>(Xb, Wqkt, bqkv, QKV);
  k_attn          <<<1024, 256, 0, stream>>>(QKV, CTX);
  k_out_gemm      <<< 512, 256, 0, stream>>>(CTX, Wct, bc, out);
}
